// LatentSDE_56470230008385
// MI455X (gfx1250) — matrix operand order repaired and checked by reading, not compiled, not run on hardware
//
#include <hip/hip_runtime.h>
#include <stdint.h>
#include <math.h>

// ---------------------------------------------------------------- constants
constexpr int D_ = 64, L_ = 128, C_ = 128, H_ = 256, T_ = 256, B_ = 512;
constexpr float DT_     = 1.0f / 255.0f;
constexpr float LOG2PI_ = 1.8378770664093453f;

typedef __bf16 bf16_t;
typedef __bf16 v16bf __attribute__((ext_vector_type(16)));
typedef __bf16 v8bf  __attribute__((ext_vector_type(8)));
typedef float  v8f   __attribute__((ext_vector_type(8)));
typedef float  v4f   __attribute__((ext_vector_type(4)));

static __device__ __forceinline__ bf16_t f2bf(float f) {
    // round-to-nearest-even f32 -> bf16
    uint32_t u = __builtin_bit_cast(uint32_t, f);
    uint32_t r = u + 0x7FFFu + ((u >> 16) & 1u);
    return __builtin_bit_cast(bf16_t, (uint16_t)(r >> 16));
}

static __device__ __forceinline__ float softplus_f(float x) {
    return fmaxf(x, 0.0f) + log1pf(expf(-fabsf(x)));
}
static __device__ __forceinline__ float sigmoid_f(float x) {
    return 1.0f / (1.0f + expf(-x));
}

// ---------------------------------------------------------------- WMMA GEMM
// Out[M,N] = act( A[M,K] * W[N,K]^T + bias[N] ),  f32 in/out, bf16 WMMA core.
// A can be a split (concat) of A0 [.,ksplit) and A1 [ksplit,K).
// Requirements: M % 64 == 0, N % 64 == 0, K % 32 == 0, ksplit % 8 == 0,
//               all pointers / leading dims 16B aligned.
// Block: 256 threads = 8 waves; block tile 64(M) x 64(N); wave tile 16x32
// (one A fragment reused across two B fragments -> 2 WMMAs per K step).
// act: 0 = identity, 1 = softplus.
__global__ __launch_bounds__(256) void gemm_bf16_kernel(
    const float* __restrict__ A0, int lda0,
    const float* __restrict__ A1, int lda1, int ksplit,
    const float* __restrict__ W, int K,
    const float* __restrict__ bias,
    float* __restrict__ Out, int N, int act)
{
    __shared__ __align__(16) bf16_t ldsA[64][40]; // +8 pad: keep 16B align, break conflicts
    __shared__ __align__(16) bf16_t ldsW[64][40];

    const int tid  = threadIdx.x;
    const int lane = tid & 31;
    const int wid  = tid >> 5;
    const int wm   = wid >> 1;          // 0..3  (M sub-tile, 16 rows)
    const int wn   = wid & 1;           // 0..1  (N sub-tile, 32 cols)
    const int m    = lane & 15;         // row (A) / col (B) within 16-wide tile
    const int half = (lane >> 4) & 1;   // half-wave select

    const long m0 = (long)blockIdx.y * 64;
    const int  n0 = blockIdx.x * 64;

    // stagers: 64 rows x 32 cols each, 8 floats (2 x b128) per thread
    const int srow = tid >> 2, scc = (tid & 3) * 8;

    v8f acc0 = {}, acc1 = {};

    for (int k0 = 0; k0 < K; k0 += 32) {
        __syncthreads();
        { // stage A tile (f32 -> bf16). ksplit is a multiple of 8 in all uses,
          // so an 8-wide chunk never straddles the concat boundary.
            const long grow = m0 + srow;
            const int  kk   = k0 + scc;
            const float* src = (kk < ksplit) ? (A0 + grow * (long)lda0 + kk)
                                             : (A1 + grow * (long)lda1 + (kk - ksplit));
            v4f p0 = *(const v4f*)(src);
            v4f p1 = *(const v4f*)(src + 4);
            if (k0 + 32 < K) __builtin_prefetch(src + 32, 0, 1); // global_prefetch_b8
            #pragma unroll
            for (int j = 0; j < 4; ++j) {
                ldsA[srow][scc + j]     = f2bf(p0[j]);
                ldsA[srow][scc + 4 + j] = f2bf(p1[j]);
            }
        }
        { // stage W tile
            const float* src = W + (long)(n0 + srow) * K + k0 + scc;
            v4f p0 = *(const v4f*)(src);
            v4f p1 = *(const v4f*)(src + 4);
            if (k0 + 32 < K) __builtin_prefetch(src + 32, 0, 1);
            #pragma unroll
            for (int j = 0; j < 4; ++j) {
                ldsW[srow][scc + j]     = f2bf(p0[j]);
                ldsW[srow][scc + 4 + j] = f2bf(p1[j]);
            }
        }
        __syncthreads();

        // Build fragments in contiguous k order, shared by A and B.
        // A (16x32): lane holds row m; K = 16*half + i -> one contiguous 16 run.
        // B (32x16): lane holds col m; K = 16*half + i -> one contiguous 16 run.
        v16bf a, b0, b1;
        {
            const v8bf* pa0 = (const v8bf*)&ldsA[wm * 16 + m][half * 16];
            const v8bf* pa1 = (const v8bf*)&ldsA[wm * 16 + m][half * 16 + 8];
            const v8bf* pb00 = (const v8bf*)&ldsW[wn * 32 + m][half * 16];
            const v8bf* pb01 = (const v8bf*)&ldsW[wn * 32 + m][half * 16 + 8];
            const v8bf* pb10 = (const v8bf*)&ldsW[wn * 32 + 16 + m][half * 16];
            const v8bf* pb11 = (const v8bf*)&ldsW[wn * 32 + 16 + m][half * 16 + 8];
            #pragma unroll
            for (int j = 0; j < 8; ++j) {
                a[j]  = (*pa0)[j];  a[8 + j]  = (*pa1)[j];
                b0[j] = (*pb00)[j]; b0[8 + j] = (*pb01)[j];
                b1[j] = (*pb10)[j]; b1[8 + j] = (*pb11)[j];
            }
        }
        acc0 = __builtin_amdgcn_wmma_f32_16x16x32_bf16(
                   false, a, false, b0, (short)0, acc0, false, false);
        acc1 = __builtin_amdgcn_wmma_f32_16x16x32_bf16(
                   false, a, false, b1, (short)0, acc1, false, false);
    }

    // Epilogue. C/D layout: VGPR r -> row (r + 8*half), col = lane&15.
    const int  col0 = n0 + wn * 32 + m;
    const int  col1 = col0 + 16;
    const float bv0 = bias[col0];
    const float bv1 = bias[col1];
    const long rowbase = m0 + wm * 16 + half * 8;
    #pragma unroll
    for (int r = 0; r < 8; ++r) {
        float v0 = acc0[r] + bv0;
        float v1 = acc1[r] + bv1;
        if (act == 1) { v0 = softplus_f(v0); v1 = softplus_f(v1); }
        Out[(rowbase + r) * (long)N + col0] = v0;
        Out[(rowbase + r) * (long)N + col1] = v1;
    }
}

static inline void gemm(hipStream_t s, const float* A0, int lda0,
                        const float* W, int K, const float* bias,
                        float* Out, int M, int N, int act,
                        const float* A1 = nullptr, int lda1 = 0, int ksplit = -1)
{
    if (!A1) { A1 = A0; lda1 = lda0; ksplit = K; }
    dim3 grid(N / 64, M / 64);
    gemm_bf16_kernel<<<grid, 256, 0, s>>>(A0, lda0, A1, lda1, ksplit, W, K, bias, Out, N, act);
}

// ---------------------------------------------------------------- GRU pieces
__global__ void init_kernel(float* h, float* qpath) {
    int i = blockIdx.x * blockDim.x + threadIdx.x;
    if (i < B_ * H_) h[i] = 0.0f;
    if (i < B_)      qpath[i] = 0.0f;
}

__global__ void gru_gate_kernel(const float* __restrict__ Gx,
                                const float* __restrict__ Gh,
                                float* __restrict__ h,
                                float* __restrict__ ctx_h_t)
{
    int idx = blockIdx.x * blockDim.x + threadIdx.x;     // B*H
    int b = idx >> 8, j = idx & 255;                     // H = 256
    const float* gx = Gx + (long)b * 768;
    const float* gh = Gh + (long)b * 768;
    float r = sigmoid_f(gx[j] + gh[j]);
    float z = sigmoid_f(gx[256 + j] + gh[256 + j]);
    float n = tanhf(gx[512 + j] + r * gh[512 + j]);
    float hn = (1.0f - z) * n + z * h[idx];
    h[idx] = hn;
    ctx_h_t[idx] = hn;
}

// ---------------------------------------------------------------- z0 / KL
__global__ void z0_kernel(const float* __restrict__ qbuf, const float* __restrict__ eps,
                          const float* __restrict__ pm, const float* __restrict__ pls,
                          float* __restrict__ y, float* __restrict__ zs0,
                          float* __restrict__ q0p)
{
    int b = blockIdx.x, l = threadIdx.x;                 // L = 128
    float qm  = qbuf[b * 256 + l];
    float qls = qbuf[b * 256 + 128 + l];
    float z0  = qm + expf(qls) * eps[b * 128 + l];
    y[b * 128 + l] = z0;
    zs0[b * 128 + l] = z0;
    float pmv = pm[l], plsv = pls[l];
    float dm = qm - pmv;
    float kl = plsv - qls + (expf(2.0f * qls) + dm * dm) / (2.0f * expf(2.0f * plsv)) - 0.5f;
    __shared__ float red[128];
    red[l] = kl; __syncthreads();
    for (int s = 64; s > 0; s >>= 1) { if (l < s) red[l] += red[l + s]; __syncthreads(); }
    if (l == 0) q0p[b] = red[0];
}

// ------------------------------------------------- diagonal noise net g(.)
// Computes gy = g(y) and gmid = 0.5*(g(y)+g(y+gy*dw))*dw  (Euler-Heun midpoint)
__global__ void g_kernel(const float* __restrict__ y, const float* __restrict__ dWrow,
                         const float* __restrict__ w1, const float* __restrict__ b1,
                         const float* __restrict__ w2, const float* __restrict__ b2,
                         float* __restrict__ gy_out, float* __restrict__ gmid_out,
                         float sqdt)
{
    int gt = blockIdx.x * blockDim.x + threadIdx.x;      // B*L
    int l = gt & 127;
    float yv = y[gt];
    const float* W1 = w1 + (long)l * H_;
    const float* B1 = b1 + (long)l * H_;
    const float* W2 = w2 + (long)l * H_;
    float acc = 0.0f;
    for (int hh = 0; hh < H_; ++hh)
        acc += softplus_f(yv * W1[hh] + B1[hh]) * W2[hh];
    float gy = sigmoid_f(acc + b2[l]);
    float dw = dWrow[gt] * sqdt;
    float yt = yv + gy * dw;
    float acc2 = 0.0f;
    for (int hh = 0; hh < H_; ++hh)
        acc2 += softplus_f(yt * W1[hh] + B1[hh]) * W2[hh];
    float g2 = sigmoid_f(acc2 + b2[l]);
    gy_out[gt]   = gy;
    gmid_out[gt] = 0.5f * (gy + g2) * dw;
}

// ----------------------------------------------------- SDE state update
__global__ void sde_update_kernel(float* __restrict__ y, const float* __restrict__ fy,
                                  const float* __restrict__ hy, const float* __restrict__ gy,
                                  const float* __restrict__ gmid, float* __restrict__ zs_next,
                                  float* __restrict__ qpath, float dt)
{
    int b = blockIdx.x, l = threadIdx.x;                 // L = 128
    int idx = b * 128 + l;
    float f = fy[idx], h = hy[idx];
    float yn = y[idx] + f * dt + gmid[idx];
    y[idx] = yn;
    zs_next[idx] = yn;
    float u = (f - h) / gy[idx];
    __shared__ float red[128];
    red[l] = u * u; __syncthreads();
    for (int s = 64; s > 0; s >>= 1) { if (l < s) red[l] += red[l + s]; __syncthreads(); }
    if (l == 0) qpath[b] += 0.5f * red[0] * dt;
}

// ----------------------------------------------------- likelihood
__global__ void lik_kernel(const float* __restrict__ xs, const float* __restrict__ xh,
                           const float* __restrict__ sc, float* __restrict__ lp_partial)
{
    long tb = blockIdx.x;                                // T*B blocks
    int d = threadIdx.x;                                 // D = 64
    long i = tb * 64 + d;
    float s = sc[i];
    float r = (xs[i] - xh[i]) / s;
    float lp = -0.5f * r * r - logf(s) - 0.5f * LOG2PI_;
    __shared__ float red[64];
    red[d] = lp; __syncthreads();
    for (int st = 32; st > 0; st >>= 1) { if (d < st) red[d] += red[d + st]; __syncthreads(); }
    if (d == 0) lp_partial[tb] = red[0];
}

__global__ void finalize_kernel(const float* __restrict__ lp_partial,
                                const float* __restrict__ q0p,
                                const float* __restrict__ qpath,
                                float* __restrict__ out)
{
    __shared__ float red[256];
    int t = threadIdx.x;
    float s = 0.0f;
    for (long i = t; i < (long)T_ * B_; i += 256) s += lp_partial[i];
    red[t] = s; __syncthreads();
    for (int st = 128; st > 0; st >>= 1) { if (t < st) red[t] += red[t + st]; __syncthreads(); }
    float lpxs = red[0] / (float)B_;
    __syncthreads();
    float s2 = 0.0f;
    for (int i = t; i < B_; i += 256) s2 += q0p[i] + qpath[i];
    red[t] = s2; __syncthreads();
    for (int st = 128; st > 0; st >>= 1) { if (t < st) red[t] += red[t + st]; __syncthreads(); }
    if (t == 0) { out[0] = lpxs; out[1] = red[0] / (float)B_; }
}

// ---------------------------------------------------------------- driver
extern "C" void kernel_launch(void* const* d_in, const int* in_sizes, int n_in,
                              void* d_out, int out_size, void* d_ws, size_t ws_size,
                              hipStream_t stream)
{
    // setup_inputs() insertion order, params flattened recursively, (W,b) pairs in order.
    const float* xs      = (const float*)d_in[0];
    // d_in[1] = ts (implicit: searchsorted(ts, ts[i], 'right') == i+1)
    const float* eps_z0  = (const float*)d_in[2];
    const float* dW      = (const float*)d_in[3];
    // d_in[4] = noise_std (unused by reference math)
    const float* gru_w_ih = (const float*)d_in[5];
    const float* gru_w_hh = (const float*)d_in[6];
    const float* gru_b_ih = (const float*)d_in[7];
    const float* gru_b_hh = (const float*)d_in[8];
    const float* encW = (const float*)d_in[9];
    const float* encB = (const float*)d_in[10];
    const float* qzW  = (const float*)d_in[11];
    const float* qzB  = (const float*)d_in[12];
    const float* fW[4] = {(const float*)d_in[13], (const float*)d_in[15],
                          (const float*)d_in[17], (const float*)d_in[19]};
    const float* fB[4] = {(const float*)d_in[14], (const float*)d_in[16],
                          (const float*)d_in[18], (const float*)d_in[20]};
    const float* hW[4] = {(const float*)d_in[21], (const float*)d_in[23],
                          (const float*)d_in[25], (const float*)d_in[27]};
    const float* hB[4] = {(const float*)d_in[22], (const float*)d_in[24],
                          (const float*)d_in[26], (const float*)d_in[28]};
    const float* g_w1 = (const float*)d_in[29];
    const float* g_b1 = (const float*)d_in[30];
    const float* g_w2 = (const float*)d_in[31];
    const float* g_b2 = (const float*)d_in[32];
    const float* pW[2] = {(const float*)d_in[33], (const float*)d_in[35]};
    const float* pB[2] = {(const float*)d_in[34], (const float*)d_in[36]};
    const float* nW   = (const float*)d_in[37];
    const float* nB   = (const float*)d_in[38];
    const float* pz0m  = (const float*)d_in[39];
    const float* pz0ls = (const float*)d_in[40];

    float* ws = (float*)d_ws;

    // workspace layout (float offsets)
    constexpr size_t O_CTXH = 0;                                   // T*B*H
    constexpr size_t O_CTXL = O_CTXH + (size_t)T_ * B_ * H_;       // T*B*C
    constexpr size_t O_ZS   = O_CTXL + (size_t)T_ * B_ * C_;       // T*B*L
    constexpr size_t O_GX   = O_ZS   + (size_t)T_ * B_ * L_;       // B*3H
    constexpr size_t O_GH   = O_GX   + (size_t)B_ * 3 * H_;        // B*3H
    constexpr size_t O_HST  = O_GH   + (size_t)B_ * 3 * H_;        // B*H
    constexpr size_t O_Y    = O_HST  + (size_t)B_ * H_;            // B*L
    constexpr size_t O_QB   = O_Y    + (size_t)B_ * L_;            // B*2L
    constexpr size_t O_BA   = O_QB   + (size_t)B_ * 2 * L_;        // B*512
    constexpr size_t O_BB   = O_BA   + (size_t)B_ * 512;           // B*512
    constexpr size_t O_FY   = O_BB   + (size_t)B_ * 512;           // B*L
    constexpr size_t O_HY   = O_FY   + (size_t)B_ * L_;            // B*L
    constexpr size_t O_GY   = O_HY   + (size_t)B_ * L_;            // B*L
    constexpr size_t O_GM   = O_GY   + (size_t)B_ * L_;            // B*L
    constexpr size_t O_QP   = O_GM   + (size_t)B_ * L_;            // B
    constexpr size_t O_Q0   = O_QP   + B_;                         // B

    // decoder aliases reuse the GRU-state region (free after enc_lin GEMM)
    float* bufP  = ws + O_CTXH;                         // T*B*2D
    float* xsh   = bufP + (size_t)T_ * B_ * 2 * D_;     // T*B*D
    float* scale = xsh  + (size_t)T_ * B_ * D_;         // T*B*D
    float* lp_partial = ws + O_GX;                      // T*B (reuses Gx scratch)

    // ---- init (h = 0, qpath = 0)
    init_kernel<<<dim3((B_ * H_) / 256), 256, 0, stream>>>(ws + O_HST, ws + O_QP);

    // ---- encoder GRU over time-reversed xs; store state at original index t
    for (int j = 0; j < T_; ++j) {
        int t = T_ - 1 - j;
        gemm(stream, xs + (size_t)t * B_ * D_, D_, gru_w_ih, D_, gru_b_ih,
             ws + O_GX, B_, 3 * H_, 0);
        gemm(stream, ws + O_HST, H_, gru_w_hh, H_, gru_b_hh,
             ws + O_GH, B_, 3 * H_, 0);
        gru_gate_kernel<<<dim3((B_ * H_) / 256), 256, 0, stream>>>(
            ws + O_GX, ws + O_GH, ws + O_HST, ws + O_CTXH + (size_t)t * B_ * H_);
    }

    // ---- one big context projection: [T*B, H] x [H -> C]
    gemm(stream, ws + O_CTXH, H_, encW, H_, encB, ws + O_CTXL, T_ * B_, C_, 0);

    // ---- posterior z0
    gemm(stream, ws + O_CTXL, C_, qzW, C_, qzB, ws + O_QB, B_, 2 * L_, 0);
    z0_kernel<<<dim3(B_), dim3(L_), 0, stream>>>(
        ws + O_QB, eps_z0, pz0m, pz0ls, ws + O_Y, ws + O_ZS, ws + O_Q0);

    // ---- SDE scan (Stratonovich Euler-Heun, ctx index = i+1)
    const float sqdt = sqrtf(DT_);
    for (int i = 0; i < T_ - 1; ++i) {
        const float* cctx = ws + O_CTXL + (size_t)(i + 1) * B_ * C_;
        // f MLP on concat([y, c]) -> fy
        gemm(stream, ws + O_Y, L_, fW[0], L_ + C_, fB[0], ws + O_BA, B_, H_, 1,
             cctx, C_, L_);
        gemm(stream, ws + O_BA, H_,     fW[1], H_,     fB[1], ws + O_BB, B_, 2 * H_, 1);
        gemm(stream, ws + O_BB, 2 * H_, fW[2], 2 * H_, fB[2], ws + O_BA, B_, H_,     1);
        gemm(stream, ws + O_BA, H_,     fW[3], H_,     fB[3], ws + O_FY, B_, L_,     0);
        // h MLP on y -> hy
        gemm(stream, ws + O_Y,  L_,     hW[0], L_,     hB[0], ws + O_BA, B_, H_,     1);
        gemm(stream, ws + O_BA, H_,     hW[1], H_,     hB[1], ws + O_BB, B_, 2 * H_, 1);
        gemm(stream, ws + O_BB, 2 * H_, hW[2], 2 * H_, hB[2], ws + O_BA, B_, H_,     1);
        gemm(stream, ws + O_BA, H_,     hW[3], H_,     hB[3], ws + O_HY, B_, L_,     0);
        // diagonal diffusion + Heun midpoint
        g_kernel<<<dim3((B_ * L_) / 256), 256, 0, stream>>>(
            ws + O_Y, dW + (size_t)i * B_ * L_, g_w1, g_b1, g_w2, g_b2,
            ws + O_GY, ws + O_GM, sqdt);
        // state update + logqp path increment
        sde_update_kernel<<<dim3(B_), dim3(L_), 0, stream>>>(
            ws + O_Y, ws + O_FY, ws + O_HY, ws + O_GY, ws + O_GM,
            ws + O_ZS + (size_t)(i + 1) * B_ * L_, ws + O_QP, DT_);
    }

    // ---- decoder on all T*B rows
    gemm(stream, ws + O_ZS, L_, pW[0], L_,     pB[0], bufP,  T_ * B_, 2 * D_, 1);
    gemm(stream, bufP, 2 * D_,  pW[1], 2 * D_, pB[1], xsh,   T_ * B_, D_,     0);
    gemm(stream, ws + O_ZS, L_, nW,    L_,     nB,    scale, T_ * B_, D_,     1);

    // ---- likelihood + final reductions
    lik_kernel<<<dim3(T_ * B_), dim3(D_), 0, stream>>>(xs, xsh, scale, lp_partial);
    finalize_kernel<<<dim3(1), dim3(256), 0, stream>>>(
        lp_partial, ws + O_Q0, ws + O_QP, (float*)d_out);
}
